// OutputLayer_39762807226529
// MI455X (gfx1250) — compile-verified, matrix operand order checked by reading, not run on hardware
//
#include <hip/hip_runtime.h>
#include <hip/hip_bf16.h>
#include <math.h>

typedef __attribute__((ext_vector_type(16))) __bf16 v16bf;
typedef __attribute__((ext_vector_type(8)))  float  v8f;
typedef unsigned short u16;
typedef unsigned int   u32;

static_assert(sizeof(v16bf) == 32, "v16bf must be 32 bytes");

// ---------------------------------------------------------------------------
// helpers
// ---------------------------------------------------------------------------
__device__ __forceinline__ u16 f2bf(float f) {
  union { float f; u32 u; } x; x.f = f;
  u32 r = x.u + 0x7FFFu + ((x.u >> 16) & 1u);   // round-to-nearest-even
  return (u16)(r >> 16);
}
__device__ __forceinline__ float gelu_f(float x) {
  return 0.5f * x * (1.0f + erff(x * 0.70710678118654752f));
}

union FragU { uint4 u[2]; v16bf v; };

// B fragment: pre-swizzled global buffer, 512 bf16 per fragment,
// lane-contiguous 16 bf16 (32B) per lane.
__device__ __forceinline__ v16bf load_bfrag(const u16* __restrict__ w, int frag, int lane) {
  FragU x;
  const uint4* p = (const uint4*)(w + ((size_t)frag * 32 + lane) * 16);
  x.u[0] = p[0];
  x.u[1] = p[1];
  return x.v;
}

// A fragment from LDS row (bf16, row-major [M][K] with padded stride).
// 16-bit A 16x32 layout: half 0 lanes: K = kb+0..7 (v0-3), kb+16..23 (v4-7);
// half 1 lanes: K = kb+8..15, kb+24..31.
__device__ __forceinline__ v16bf load_afrag(const u16* lds_row, int kofs, int hf) {
  FragU x;
  x.u[0] = *(const uint4*)(lds_row + kofs + hf * 8);
  x.u[1] = *(const uint4*)(lds_row + kofs + 16 + hf * 8);
  return x.v;
}

#define WMMA_BF16(a, b, c) \
  __builtin_amdgcn_wmma_f32_16x16x32_bf16(false, (a), false, (b), (short)0, (c), false, false)

// Same-wave cross-lane LDS ordering: wait on CDNA5 split DS counter.
#define LDS_FENCE() asm volatile("s_wait_dscnt 0" ::: "memory")

static constexpr int XST = 136;  // 128 + 8 bf16 row pad (272B stride, conflict-free)
static constexpr int GST = 40;   // 32 + 8 bf16 row pad (80B stride, conflict-free)

// Write C-layout accumulators (f32) into LDS as bf16 A-layout [16][GST],
// optional bias + exact GELU.
__device__ __forceinline__ void c_to_lds(u16* G, const v8f& c0, const v8f& c1,
                                         const float* __restrict__ bias, bool act,
                                         int lane) {
  int nlo = lane & 15, hf = lane >> 4;
  float b0v = bias ? bias[nlo] : 0.0f;
  float b1v = bias ? bias[16 + nlo] : 0.0f;
#pragma unroll
  for (int r = 0; r < 8; ++r) {
    int M = r + 8 * hf;
    float v0 = c0[r] + b0v;
    float v1 = c1[r] + b1v;
    if (act) { v0 = gelu_f(v0); v1 = gelu_f(v1); }
    G[M * GST + nlo]      = f2bf(v0);
    G[M * GST + 16 + nlo] = f2bf(v1);
  }
}

// ---------------------------------------------------------------------------
// weight swizzle: f32 W[K][N] row-major -> bf16 B-fragment layout
// fragment f = kbi*(Npad/16)+nb ; element j of lane l mirrors A-layout duality
// ---------------------------------------------------------------------------
__global__ void swizzle_kernel(const float* __restrict__ W, u16* __restrict__ dst,
                               int K, int N, int Npad) {
  int tid = blockIdx.x * blockDim.x + threadIdx.x;
  int total = (K / 32) * (Npad / 16) * 512;
  if (tid >= total) return;
  int j = tid & 15;
  int lane = (tid >> 4) & 31;
  int f = tid >> 9;
  int NB = Npad / 16;
  int kbi = f / NB, nb = f % NB;
  int hf = lane >> 4, nlo = lane & 15;
  int k = kbi * 32 + (j < 8 ? hf * 8 + j : 16 + hf * 8 + (j - 8));
  int n = nb * 16 + nlo;
  float v = (n < N) ? W[(size_t)k * N + n] : 0.0f;
  dst[tid] = f2bf(v);
}

// ---------------------------------------------------------------------------
// Atom terms: out MLP [128 -> 32 -> 32 -> OUT], input rows of x_h directly.
// One wave handles a 16-row tile.
// ---------------------------------------------------------------------------
template <int OUT_DIM, int WAVES>
__global__ void __launch_bounds__(WAVES * 32)
atom_kernel(const float* __restrict__ xh, int n_rows,
            const u16* __restrict__ w0, const float* __restrict__ b0,
            const u16* __restrict__ w1, const float* __restrict__ b1,
            const u16* __restrict__ w2, const float* __restrict__ b2,
            float* __restrict__ out) {
  __shared__ __align__(16) u16 Xs[WAVES][16][XST];
  __shared__ __align__(16) u16 Gs[WAVES][16][GST];
  int wv = threadIdx.x >> 5, lane = threadIdx.x & 31;
  long tile = (long)blockIdx.x * WAVES + wv;
  long base = tile * 16;
  if (base >= n_rows) return;
  // stage 16 rows, f32 -> bf16
  for (int m = 0; m < 16; ++m) {
    long row = base + m; if (row >= n_rows) row = n_rows - 1;
    const float4 f = *(const float4*)(xh + (size_t)row * 128 + lane * 4);
    u32 lo = (u32)f2bf(f.x) | ((u32)f2bf(f.y) << 16);
    u32 hi = (u32)f2bf(f.z) | ((u32)f2bf(f.w) << 16);
    *(uint2*)&Xs[wv][m][lane * 4] = make_uint2(lo, hi);
  }
  LDS_FENCE();
  int nlo = lane & 15, hf = lane >> 4;
  u16* G = &Gs[wv][0][0];
  // layer0: 128 -> 32, GELU
  v8f c0 = {}, c1 = {};
#pragma unroll
  for (int kb = 0; kb < 128; kb += 32) {
    v16bf a = load_afrag(&Xs[wv][nlo][0], kb, hf);
    int kbi = kb >> 5;
    c0 = WMMA_BF16(a, load_bfrag(w0, kbi * 2 + 0, lane), c0);
    c1 = WMMA_BF16(a, load_bfrag(w0, kbi * 2 + 1, lane), c1);
  }
  c_to_lds(G, c0, c1, b0, true, lane);
  LDS_FENCE();
  // layer1: 32 -> 32, GELU
  {
    v16bf a = load_afrag(&G[nlo * GST], 0, hf);
    v8f d0 = {}, d1 = {};
    d0 = WMMA_BF16(a, load_bfrag(w1, 0, lane), d0);
    d1 = WMMA_BF16(a, load_bfrag(w1, 1, lane), d1);
    c_to_lds(G, d0, d1, b1, true, lane);
    LDS_FENCE();
  }
  // layer2: 32 -> OUT (padded to 16), 15*tanh
  {
    v16bf a = load_afrag(&G[nlo * GST], 0, hf);
    v8f d0 = {};
    d0 = WMMA_BF16(a, load_bfrag(w2, 0, lane), d0);
    if (nlo < OUT_DIM) {
      float bb = b2[nlo];
#pragma unroll
      for (int r = 0; r < 8; ++r) {
        long row = base + r + 8 * hf;
        if (row < n_rows)
          out[row * OUT_DIM + nlo] = 15.0f * tanhf(d0[r] + bb);
      }
    }
  }
}

// ---------------------------------------------------------------------------
// Symmetric terms: sum over permutations of post MLP [NNODE*128->32->32->32],
// then out MLP [32->32->OUT], 15*tanh.
// IMPROPER: perms = (node0, node_{p+1}); else perm0 = identity, perm1 = reversed.
// ---------------------------------------------------------------------------
template <int NIDX, int NNODE, int NPERM, int OUT_DIM, bool IMPROPER, int WAVES>
__global__ void __launch_bounds__(WAVES * 32)
sym_kernel(const float* __restrict__ xh, const int* __restrict__ idx, int n_rows,
           const u16* __restrict__ w0, const float* __restrict__ b0,
           const u16* __restrict__ w1, const float* __restrict__ b1,
           const u16* __restrict__ w2, const float* __restrict__ b2,
           const u16* __restrict__ w3, const float* __restrict__ b3,
           const u16* __restrict__ w4, const float* __restrict__ b4,
           float* __restrict__ out) {
  constexpr int K0 = NNODE * 128;
  __shared__ __align__(16) u16 Xs[WAVES][NIDX][16][XST];
  __shared__ __align__(16) u16 Gs[WAVES][16][GST];
  int wv = threadIdx.x >> 5, lane = threadIdx.x & 31;
  long tile = (long)blockIdx.x * WAVES + wv;
  long base = tile * 16;
  if (base >= n_rows) return;
  // stage all referenced node blocks once (re-used by every permutation)
  for (int nd = 0; nd < NIDX; ++nd) {
    for (int m = 0; m < 16; ++m) {
      long row = base + m; if (row >= n_rows) row = n_rows - 1;
      int node = idx[(size_t)nd * n_rows + row];
      const float4 f = *(const float4*)(xh + (size_t)node * 128 + lane * 4);
      u32 lo = (u32)f2bf(f.x) | ((u32)f2bf(f.y) << 16);
      u32 hi = (u32)f2bf(f.z) | ((u32)f2bf(f.w) << 16);
      *(uint2*)&Xs[wv][nd][m][lane * 4] = make_uint2(lo, hi);
    }
  }
  LDS_FENCE();
  int nlo = lane & 15, hf = lane >> 4;
  u16* G = &Gs[wv][0][0];
  v8f s0 = {}, s1 = {};  // permutation-summed post-MLP output (bias included)
  for (int p = 0; p < NPERM; ++p) {
    // post layer0: K0 -> 32, GELU
    v8f c0 = {}, c1 = {};
#pragma unroll
    for (int kb = 0; kb < K0; kb += 32) {
      int slot = kb >> 7;
      int node = IMPROPER ? (slot == 0 ? 0 : p + 1)
                          : (p == 0 ? slot : NNODE - 1 - slot);
      v16bf a = load_afrag(&Xs[wv][node][nlo][0], kb & 127, hf);
      int kbi = kb >> 5;
      c0 = WMMA_BF16(a, load_bfrag(w0, kbi * 2 + 0, lane), c0);
      c1 = WMMA_BF16(a, load_bfrag(w0, kbi * 2 + 1, lane), c1);
    }
    c_to_lds(G, c0, c1, b0, true, lane);
    LDS_FENCE();
    // post layer1: 32 -> 32, GELU
    {
      v16bf a = load_afrag(&G[nlo * GST], 0, hf);
      v8f d0 = {}, d1 = {};
      d0 = WMMA_BF16(a, load_bfrag(w1, 0, lane), d0);
      d1 = WMMA_BF16(a, load_bfrag(w1, 1, lane), d1);
      c_to_lds(G, d0, d1, b1, true, lane);
      LDS_FENCE();
    }
    // post layer2: 32 -> 32 (no activation), accumulate over perms
    {
      v16bf a = load_afrag(&G[nlo * GST], 0, hf);
      v8f d0 = {}, d1 = {};
      d0 = WMMA_BF16(a, load_bfrag(w2, 0, lane), d0);
      d1 = WMMA_BF16(a, load_bfrag(w2, 1, lane), d1);
      float bb0 = b2[nlo], bb1 = b2[16 + nlo];
#pragma unroll
      for (int r = 0; r < 8; ++r) { s0[r] += d0[r] + bb0; s1[r] += d1[r] + bb1; }
    }
  }
  // out layer0: 32 -> 32, GELU
  c_to_lds(G, s0, s1, nullptr, false, lane);
  LDS_FENCE();
  {
    v16bf a = load_afrag(&G[nlo * GST], 0, hf);
    v8f d0 = {}, d1 = {};
    d0 = WMMA_BF16(a, load_bfrag(w3, 0, lane), d0);
    d1 = WMMA_BF16(a, load_bfrag(w3, 1, lane), d1);
    c_to_lds(G, d0, d1, b3, true, lane);
    LDS_FENCE();
  }
  // out layer1: 32 -> OUT (padded to 16), 15*tanh
  {
    v16bf a = load_afrag(&G[nlo * GST], 0, hf);
    v8f d0 = {};
    d0 = WMMA_BF16(a, load_bfrag(w4, 0, lane), d0);
    if (nlo < OUT_DIM) {
      float bb = b4[nlo];
#pragma unroll
      for (int r = 0; r < 8; ++r) {
        long row = base + r + 8 * hf;
        if (row < n_rows)
          out[row * OUT_DIM + nlo] = 15.0f * tanhf(d0[r] + bb);
      }
    }
  }
}

// ---------------------------------------------------------------------------
// host launch
// ---------------------------------------------------------------------------
extern "C" void kernel_launch(void* const* d_in, const int* in_sizes, int n_in,
                              void* d_out, int out_size, void* d_ws, size_t ws_size,
                              hipStream_t stream) {
  (void)in_sizes; (void)n_in; (void)out_size; (void)ws_size;
  constexpr int N_ATOM = 250000, N_BOND = 250000, N_ANGLE = 400000,
                N_PROPER = 300000, N_IMPROPER = 80000, PT = 4;

  const float* xh        = (const float*)d_in[0];
  const int*   bond_idx  = (const int*)d_in[3];
  const int*   angle_idx = (const int*)d_in[4];
  const int*   prop_idx  = (const int*)d_in[5];
  const int*   impr_idx  = (const int*)d_in[6];
  float* out = (float*)d_out;

  // flattened param bases (insertion order; post=None skipped for atom terms)
  const int P_SIGMA = 7, P_EPS = 13, P_CHG = 19;       // out: W,b x3
  const int P_BK = 25, P_BL = 35, P_AK = 45, P_AT = 55, P_PT = 65, P_IT = 75;

  u16* ws = (u16*)d_ws;
  size_t cur = 0;
  auto swz = [&](int src, int K, int N, int Npad) -> const u16* {
    size_t elems = (size_t)(K / 32) * (Npad / 16) * 512;
    u16* dst = ws + cur;
    cur += elems;
    int total = (int)elems;
    swizzle_kernel<<<(total + 255) / 256, 256, 0, stream>>>(
        (const float*)d_in[src], dst, K, N, Npad);
    return dst;
  };
  auto B = [&](int i) { return (const float*)d_in[i]; };

  // ---- atom terms ----
  auto run_atom = [&](int pb, float* o) {
    const u16* w0 = swz(pb + 0, 128, 32, 32);
    const u16* w1 = swz(pb + 2, 32, 32, 32);
    const u16* w2 = swz(pb + 4, 32, 1, 16);
    int tiles = (N_ATOM + 15) / 16, blocks = (tiles + 3) / 4;
    atom_kernel<1, 4><<<blocks, 128, 0, stream>>>(
        xh, N_ATOM, w0, B(pb + 1), w1, B(pb + 3), w2, B(pb + 5), o);
  };
  run_atom(P_SIGMA, out + 0);
  run_atom(P_EPS,   out + N_ATOM);
  run_atom(P_CHG,   out + 2 * N_ATOM);

  // ---- symmetric terms ----
  // bond-like (K0=256): Bond_k, Bond_length
  auto run_bond = [&](int pb, float* o) {
    const u16* w0 = swz(pb + 0, 256, 32, 32);
    const u16* w1 = swz(pb + 2, 32, 32, 32);
    const u16* w2 = swz(pb + 4, 32, 32, 32);
    const u16* w3 = swz(pb + 6, 32, 32, 32);
    const u16* w4 = swz(pb + 8, 32, 1, 16);
    int tiles = (N_BOND + 15) / 16, blocks = (tiles + 3) / 4;
    sym_kernel<2, 2, 2, 1, false, 4><<<blocks, 128, 0, stream>>>(
        xh, bond_idx, N_BOND, w0, B(pb + 1), w1, B(pb + 3), w2, B(pb + 5),
        w3, B(pb + 7), w4, B(pb + 9), o);
  };
  run_bond(P_BK, out + 3 * N_ATOM);
  run_bond(P_BL, out + 3 * N_ATOM + N_BOND);

  // angle (K0=384): Angle_k, Angle_theta
  auto run_angle = [&](int pb, float* o) {
    const u16* w0 = swz(pb + 0, 384, 32, 32);
    const u16* w1 = swz(pb + 2, 32, 32, 32);
    const u16* w2 = swz(pb + 4, 32, 32, 32);
    const u16* w3 = swz(pb + 6, 32, 32, 32);
    const u16* w4 = swz(pb + 8, 32, 1, 16);
    int tiles = (N_ANGLE + 15) / 16, blocks = (tiles + 3) / 4;
    sym_kernel<3, 3, 2, 1, false, 4><<<blocks, 128, 0, stream>>>(
        xh, angle_idx, N_ANGLE, w0, B(pb + 1), w1, B(pb + 3), w2, B(pb + 5),
        w3, B(pb + 7), w4, B(pb + 9), o);
  };
  size_t ofs = (size_t)3 * N_ATOM + 2 * N_BOND;
  run_angle(P_AK, out + ofs);
  run_angle(P_AT, out + ofs + N_ANGLE);

  // proper torsion (K0=512, OUT=4)
  {
    int pb = P_PT;
    const u16* w0 = swz(pb + 0, 512, 32, 32);
    const u16* w1 = swz(pb + 2, 32, 32, 32);
    const u16* w2 = swz(pb + 4, 32, 32, 32);
    const u16* w3 = swz(pb + 6, 32, 32, 32);
    const u16* w4 = swz(pb + 8, 32, PT, 16);
    int tiles = (N_PROPER + 15) / 16, blocks = (tiles + 3) / 4;
    sym_kernel<4, 4, 2, PT, false, 4><<<blocks, 128, 0, stream>>>(
        xh, prop_idx, N_PROPER, w0, B(pb + 1), w1, B(pb + 3), w2, B(pb + 5),
        w3, B(pb + 7), w4, B(pb + 9), out + ofs + 2 * N_ANGLE);
  }

  // improper torsion (pairs (0,j), K0=256, 3 perms)
  {
    int pb = P_IT;
    const u16* w0 = swz(pb + 0, 256, 32, 32);
    const u16* w1 = swz(pb + 2, 32, 32, 32);
    const u16* w2 = swz(pb + 4, 32, 32, 32);
    const u16* w3 = swz(pb + 6, 32, 32, 32);
    const u16* w4 = swz(pb + 8, 32, 1, 16);
    int tiles = (N_IMPROPER + 15) / 16, blocks = (tiles + 3) / 4;
    sym_kernel<4, 2, 3, 1, true, 4><<<blocks, 128, 0, stream>>>(
        xh, impr_idx, N_IMPROPER, w0, B(pb + 1), w1, B(pb + 3), w2, B(pb + 5),
        w3, B(pb + 7), w4, B(pb + 9), out + ofs + 2 * N_ANGLE + (size_t)PT * N_PROPER);
  }
}
